// UMNN_90211493085521
// MI455X (gfx1250) — matrix operand order repaired and checked by reading, not compiled, not run on hardware
//
#include <hip/hip_runtime.h>
#include <math.h>

typedef __attribute__((ext_vector_type(16))) _Float16 v16h;
typedef __attribute__((ext_vector_type(8)))  float    v8f;

#define S_NODES 51   // NB_STEPS = 50 -> 51 quadrature nodes
#define HID     128  // hidden width
#define TILE_M  16   // elements per wave
#define WAVES_PER_BLOCK 8

__global__ __launch_bounds__(256) void umnn_cc_kernel(
    const float* __restrict__ x,
    const float* __restrict__ W1,   // [128] (H,1) flattened
    const float* __restrict__ b1,   // [128]
    const float* __restrict__ W2,   // [128] (1,H) flattened
    const float* __restrict__ b2,   // [1]
    float* __restrict__ out,
    int n)
{
    __shared__ float s_t[S_NODES];  // (cos(s*pi/50)+1)/2
    __shared__ float s_w[S_NODES];  // CC weights

    const int tid = threadIdx.x;

    // ---- One-time CC weight/node construction (faithful to numpy build) ----
    if (tid < S_NODES) {
        const double PI = 3.14159265358979323846;
        const int s = tid;
        double sum = 0.0;
        for (int j = 0; j <= 50; ++j) {
            double Wv;
            if (j == 0)       Wv = 1.0;
            else if (j & 1)   continue;                       // odd -> 0
            else              Wv = 2.0 / (1.0 - (double)(j * j));
            double lam;
            if (s == 0)       lam = 0.5;
            else {
                lam = cos((double)j * (double)s * PI / 50.0);
                if (s == 50) lam *= 0.5;
            }
            sum += lam * Wv;
        }
        s_w[s] = (float)(sum * (2.0 / 50.0));
        s_t[s] = (float)((cos((double)s * PI / 50.0) + 1.0) * 0.5);
    }
    __syncthreads();

    // ---- Per-wave 16-element tile ----
    const int lane    = tid & 31;
    const int wave    = tid >> 5;
    const int half_id = lane >> 4;   // 0: lanes 0-15, 1: lanes 16-31
    const int col     = lane & 15;   // A-matrix row M owned by this lane
    const int tile    = blockIdx.x * WAVES_PER_BLOCK + wave;
    const int base    = tile * TILE_M;
    if (base + TILE_M > n) return;   // uniform per wave

    const float xm  = x[base + col];
    const float b2v = b2[0];

    // Per-lane W1/b1 slices in A layout (f16), W2 chunks in B layout (f16).
    // A element j of chunk c -> hidden index h = 32c + (j<8 ? 8*half+j : 16+8*half+j-8)
    // B element j of chunk c -> hidden index h = 32c + (j<8 ? 8*half+j : 16+8*half+j-8)  (same for all 16 cols)
    _Float16 w1a[4][16];
    _Float16 b1a[4][16];
    v16h     w2b[4];
#pragma unroll
    for (int c = 0; c < 4; ++c) {
        const int r0 = 32 * c + 8 * half_id;       // run 0 (8 consecutive h)
        const int r1 = r0 + 16;                    // run 1
#pragma unroll
        for (int j = 0; j < 8; ++j) {
            w1a[c][j]     = (_Float16)W1[r0 + j];
            w1a[c][8 + j] = (_Float16)W1[r1 + j];
            b1a[c][j]     = (_Float16)b1[r0 + j];
            b1a[c][8 + j] = (_Float16)b1[r1 + j];
        }
        const int bs = 32 * c + 8 * half_id;
#pragma unroll
        for (int j = 0; j < 16; ++j)
            w2b[c][j] = (_Float16)W2[bs + j + (j < 8 ? 0 : 8)];
    }

    v8f acc = {};  // CC-weighted sum per owned row (rows 8*half_id + r)

#pragma unroll 2
    for (int s = 0; s < S_NODES; ++s) {
        const float    u  = xm * s_t[s];
        const _Float16 uh = (_Float16)u;

        // Build ALL four A-tiles first (distinct registers) so the four
        // WMMAs below only chain through D->C (no A/B WAR hazard -> no v_nops),
        // and the next unrolled iteration's VALU can co-execute under them.
        v16h a0, a1, a2, a3;
#pragma unroll
        for (int j = 0; j < 16; ++j) {
            _Float16 v;
            v = (_Float16)(uh * w1a[0][j]) + b1a[0][j];
            a0[j] = (v > (_Float16)0.0f) ? v : (_Float16)0.0f;
            v = (_Float16)(uh * w1a[1][j]) + b1a[1][j];
            a1[j] = (v > (_Float16)0.0f) ? v : (_Float16)0.0f;
            v = (_Float16)(uh * w1a[2][j]) + b1a[2][j];
            a2[j] = (v > (_Float16)0.0f) ? v : (_Float16)0.0f;
            v = (_Float16)(uh * w1a[3][j]) + b1a[3][j];
            a3[j] = (v > (_Float16)0.0f) ? v : (_Float16)0.0f;
        }

        v8f d = {};
        d = __builtin_amdgcn_wmma_f32_16x16x32_f16(false, a0, false, w2b[0], (short)0, d, false, false);
        d = __builtin_amdgcn_wmma_f32_16x16x32_f16(false, a1, false, w2b[1], (short)0, d, false, false);
        d = __builtin_amdgcn_wmma_f32_16x16x32_f16(false, a2, false, w2b[2], (short)0, d, false, false);
        d = __builtin_amdgcn_wmma_f32_16x16x32_f16(false, a3, false, w2b[3], (short)0, d, false, false);

        // d[r] = f[row 8*half_id + r]  (replicated across the 16 columns)
        const float cw = s_w[s];
#pragma unroll
        for (int r = 0; r < 8; ++r) {
            const float y = d[r] + b2v;
            const float g = (y > 0.0f) ? (y + 1.0f) : __expf(y);   // elu(y)+1
            acc[r] = fmaf(cw, g, acc[r]);
        }
    }

    // One lane per column-half writes its 8 rows: z = acc * x * 0.5
    if (col == 0) {
#pragma unroll
        for (int r = 0; r < 8; ++r) {
            const int m = 8 * half_id + r;
            out[base + m] = acc[r] * x[base + m] * 0.5f;
        }
    }
}

extern "C" void kernel_launch(void* const* d_in, const int* in_sizes, int n_in,
                              void* d_out, int out_size, void* d_ws, size_t ws_size,
                              hipStream_t stream) {
    const float* x  = (const float*)d_in[0];
    const float* W1 = (const float*)d_in[1];
    const float* b1 = (const float*)d_in[2];
    const float* W2 = (const float*)d_in[3];
    const float* b2 = (const float*)d_in[4];
    float* out = (float*)d_out;

    const int n      = in_sizes[0];                       // 32768
    const int tiles  = (n + TILE_M - 1) / TILE_M;         // 2048
    const int blocks = (tiles + WAVES_PER_BLOCK - 1) / WAVES_PER_BLOCK;  // 256

    umnn_cc_kernel<<<blocks, 256, 0, stream>>>(x, W1, b1, W2, b2, out, n);
}
